// Multiheadattention_52278341927602
// MI455X (gfx1250) — compile-verified, matrix operand order checked by reading, not run on hardware
//
#include <hip/hip_runtime.h>
#include <math.h>

#define D_MODEL 1024
#define N_HEADS 16
#define D_K 64
#define SEQ 2048
#define BATCH 2
#define ROWS (BATCH * SEQ) /* 4096 */
#define BH (BATCH * N_HEADS) /* 32 */
#define LDSW 2052 /* padded score-row stride (floats) to avoid bank conflicts */

typedef __attribute__((ext_vector_type(16))) _Float16 v16h;
typedef __attribute__((ext_vector_type(8))) _Float16 v8h;
typedef __attribute__((ext_vector_type(8))) float v8f;
typedef __attribute__((ext_vector_type(4))) float v4f;

__device__ __forceinline__ v8f wmma16x16x32(v16h a, v16h b, v8f c) {
  // emits v_wmma_f32_16x16x32_f16
  return __builtin_amdgcn_wmma_f32_16x16x32_f16(false, a, false, b, (short)0, c,
                                                false, false);
}

// ISA fragment layout (16-bit A/B, 16x32): lane holds K {0..7,16..23} + 8*hi.
// With p already offset by (row*ld + k0 + hi*8), the fragment is two
// contiguous 16-byte groups at +0 and +16 halves.
__device__ __forceinline__ v16h load_frag_f16(const _Float16* p) {
  v8h a = *(const v8h*)(p);
  v8h b = *(const v8h*)(p + 16);
  return __builtin_shufflevector(a, b, 0, 1, 2, 3, 4, 5, 6, 7, 8, 9, 10, 11,
                                 12, 13, 14, 15);
}

// Same pattern from fp32 source (global or LDS), converting to f16.
__device__ __forceinline__ v16h cvt_frag_f32(const float* p) {
  v4f x0 = *(const v4f*)(p);
  v4f x1 = *(const v4f*)(p + 4);
  v4f x2 = *(const v4f*)(p + 16);
  v4f x3 = *(const v4f*)(p + 20);
  v16h r;
#pragma unroll
  for (int i = 0; i < 4; ++i) {
    r[i] = (_Float16)x0[i];
    r[4 + i] = (_Float16)x1[i];
    r[8 + i] = (_Float16)x2[i];
    r[12 + i] = (_Float16)x3[i];
  }
  return r;
}

// ---------------------------------------------------------------------------
// Kernel 0: convert weights fp32[K,N] -> f16[N,K] (transpose via LDS tile).
// ---------------------------------------------------------------------------
__global__ void k_wcvt(const float* __restrict__ Wq, const float* __restrict__ Wk,
                       const float* __restrict__ Wv, const float* __restrict__ Wo,
                       _Float16* __restrict__ WqT, _Float16* __restrict__ WkT,
                       _Float16* __restrict__ WvT, _Float16* __restrict__ WoT) {
  __shared__ _Float16 tile[32][33];
  const int which = blockIdx.z;
  const float* W = (which == 0) ? Wq : (which == 1) ? Wk : (which == 2) ? Wv : Wo;
  _Float16* Wt = (which == 0) ? WqT : (which == 1) ? WkT : (which == 2) ? WvT : WoT;
  const int n0 = blockIdx.x * 32, k0 = blockIdx.y * 32;
  const int tx = threadIdx.x, ty = threadIdx.y;  // 32 x 8
#pragma unroll
  for (int j = 0; j < 4; ++j)
    tile[ty + 8 * j][tx] = (_Float16)W[(size_t)(k0 + ty + 8 * j) * D_MODEL + n0 + tx];
  __syncthreads();
#pragma unroll
  for (int j = 0; j < 4; ++j)
    Wt[(size_t)(n0 + ty + 8 * j) * D_MODEL + k0 + tx] = tile[tx][ty + 8 * j];
}

// ---------------------------------------------------------------------------
// Kernel 1: fused QKV projections. Each wave: 16(M) x 64(N), 4 accumulators.
// Q/K stored [B,H,S,D_K] f16 (Q pre-scaled 1/8); V stored transposed
// [B,H,D_K,S] f16 so attn.V B-fragments are contiguous.
// ---------------------------------------------------------------------------
__global__ void k_qkv(const float* __restrict__ q, const float* __restrict__ k,
                      const float* __restrict__ v,
                      const _Float16* __restrict__ WqT,
                      const _Float16* __restrict__ WkT,
                      const _Float16* __restrict__ WvT,
                      const float* __restrict__ bq, const float* __restrict__ bk,
                      const float* __restrict__ bv, _Float16* __restrict__ qh,
                      _Float16* __restrict__ kh, _Float16* __restrict__ vhT) {
  const int w = threadIdx.x >> 5;
  const int l = threadIdx.x & 31;
  const int hi = l >> 4;
  const int ln = l & 15;
  const int row0 = blockIdx.x * 16;
  const int col0 = blockIdx.y * 512 + w * 64;
  const int which = blockIdx.z;
  const float* src = (which == 0) ? q : (which == 1) ? k : v;
  const _Float16* WT = (which == 0) ? WqT : (which == 1) ? WkT : WvT;
  const float* bias = (which == 0) ? bq : (which == 1) ? bk : bv;

  const float* ap = src + (size_t)(row0 + ln) * D_MODEL + hi * 8;
  const _Float16* bp0 = WT + (size_t)(col0 + ln) * D_MODEL + hi * 8;
  v8f acc[4] = {};
  for (int k0 = 0; k0 < D_MODEL; k0 += 32) {
    v16h a = cvt_frag_f32(ap + k0);
    __builtin_prefetch(ap + k0 + 32, 0, 3);
#pragma unroll
    for (int j = 0; j < 4; ++j) {
      const _Float16* bp = bp0 + (size_t)(j * 16) * D_MODEL + k0;
      v16h b = load_frag_f16(bp);
      __builtin_prefetch(bp + 32, 0, 3);
      acc[j] = wmma16x16x32(a, b, acc[j]);
    }
  }
  const int b = row0 >> 11;
  const int s0 = (row0 & (SEQ - 1)) + hi * 8;
  const float scale = (which == 0) ? 0.125f : 1.0f;  // 1/sqrt(64)
#pragma unroll
  for (int j = 0; j < 4; ++j) {
    const int col = col0 + j * 16 + ln;
    const int hh = col >> 6;
    const int d = col & (D_K - 1);
    const float bb = bias[col];
    if (which == 2) {  // V: transposed, contiguous 8-half store per lane
      v8h pack;
#pragma unroll
      for (int r = 0; r < 8; ++r) pack[r] = (_Float16)(acc[j][r] + bb);
      *(v8h*)(vhT + ((size_t)(b * N_HEADS + hh) * D_K + d) * SEQ + s0) = pack;
    } else {
      _Float16* dst = ((which == 0) ? qh : kh) +
                      (((size_t)(b * N_HEADS + hh) * SEQ + s0) * D_K + d);
#pragma unroll
      for (int r = 0; r < 8; ++r)
        dst[(size_t)r * D_K] = (_Float16)((acc[j][r] + bb) * scale);
    }
  }
}

// ---------------------------------------------------------------------------
// Kernel 2 (fused attention): per workgroup = (head, 16-query tile).
// Phase 1: scores = Q.K^T into 128 KB LDS (never touches HBM).
// Phase 2: softmax in LDS (wave32 shfl reductions); single coalesced write of
//          the probabilities to d_out's attn region.
// Phase 3: ctx = attn.V with A-fragments from LDS (ds_load_b128 + cvt),
//          K-dim split over the 8 waves; partials reduced via reused LDS.
// ---------------------------------------------------------------------------
__global__ void k_attn(const _Float16* __restrict__ qh,
                       const _Float16* __restrict__ kh,
                       const _Float16* __restrict__ vhT,
                       float* __restrict__ attn, _Float16* __restrict__ ctx) {
  extern __shared__ float scores[];  // [16][LDSW] fp32, reused as partials
  const int w = threadIdx.x >> 5;
  const int l = threadIdx.x & 31;
  const int hi = l >> 4;
  const int ln = l & 15;
  const int bh = blockIdx.z;
  const int sq0 = blockIdx.x * 16;

  // ---- Phase 1: Q.K^T -> LDS --------------------------------------------
  const _Float16* Qp =
      qh + (size_t)bh * SEQ * D_K + (size_t)(sq0 + ln) * D_K + hi * 8;
  const v16h a0 = load_frag_f16(Qp);
  const v16h a1 = load_frag_f16(Qp + 32);
  const _Float16* Kbase = kh + (size_t)bh * SEQ * D_K;
#pragma unroll
  for (int g = 0; g < 4; ++g) {
    const int sk0 = w * 256 + g * 64;
    v8f acc[4] = {};
#pragma unroll
    for (int j = 0; j < 4; ++j) {
      const _Float16* kp = Kbase + (size_t)(sk0 + j * 16 + ln) * D_K + hi * 8;
      acc[j] = wmma16x16x32(a0, load_frag_f16(kp), acc[j]);
      acc[j] = wmma16x16x32(a1, load_frag_f16(kp + 32), acc[j]);
    }
#pragma unroll
    for (int j = 0; j < 4; ++j)
#pragma unroll
      for (int r = 0; r < 8; ++r)
        scores[(r + hi * 8) * LDSW + sk0 + j * 16 + ln] = acc[j][r];
  }
  __syncthreads();

  // ---- Phase 2: softmax in LDS + write attn -----------------------------
#pragma unroll
  for (int rr = 0; rr < 2; ++rr) {
    const int row = w * 2 + rr;
    float* srow = scores + row * LDSW;
    float m = -INFINITY;
    for (int i = 0; i < 64; ++i) m = fmaxf(m, srow[i * 32 + l]);
#pragma unroll
    for (int mask = 16; mask > 0; mask >>= 1)
      m = fmaxf(m, __shfl_xor(m, mask, 32));
    float sum = 0.f;
    for (int i = 0; i < 64; ++i) {
      const float e = __expf(srow[i * 32 + l] - m);
      srow[i * 32 + l] = e;
      sum += e;
    }
#pragma unroll
    for (int mask = 16; mask > 0; mask >>= 1) sum += __shfl_xor(sum, mask, 32);
    const float inv = 1.0f / sum;
    float* gattn = attn + (size_t)bh * SEQ * SEQ + (size_t)(sq0 + row) * SEQ;
    for (int i = 0; i < 64; ++i) {
      const float p = srow[i * 32 + l] * inv;
      srow[i * 32 + l] = p;
      gattn[i * 32 + l] = p;
    }
  }
  __syncthreads();

  // ---- Phase 3: ctx = attn.V, K split over waves ------------------------
  v8f acc[4] = {};
  const _Float16* V = vhT + (size_t)bh * D_K * SEQ + (size_t)ln * SEQ + hi * 8;
  const int kwbase = w * 256;
#pragma unroll
  for (int kk = 0; kk < 256; kk += 32) {
    const int k0 = kwbase + kk;
    v16h a = cvt_frag_f32(scores + ln * LDSW + k0 + hi * 8);  // ds_load_b128
#pragma unroll
    for (int j = 0; j < 4; ++j)
      acc[j] = wmma16x16x32(a, load_frag_f16(V + (size_t)(j * 16) * SEQ + k0),
                            acc[j]);
  }
  __syncthreads();
  float* partial = scores;  // reuse LDS: [8 waves][16 rows][64 cols]
#pragma unroll
  for (int j = 0; j < 4; ++j)
#pragma unroll
    for (int r = 0; r < 8; ++r)
      partial[w * 1024 + (r + hi * 8) * 64 + j * 16 + ln] = acc[j][r];
  __syncthreads();
  const int b = bh >> 4, hh = bh & 15;
  for (int o = threadIdx.x; o < 1024; o += 256) {
    float s = 0.f;
#pragma unroll
    for (int ww = 0; ww < 8; ++ww) s += partial[ww * 1024 + o];
    const int row = o >> 6, col = o & 63;
    ctx[(size_t)(b * SEQ + sq0 + row) * D_MODEL + hh * D_K + col] = (_Float16)s;
  }
}

// ---------------------------------------------------------------------------
// Kernel 3: y = ctx.Wo + bo + residual(q). Each wave: 16 x 64, 4 accums.
// ---------------------------------------------------------------------------
__global__ void k_proj(const _Float16* __restrict__ ctx,
                       const _Float16* __restrict__ WoT,
                       const float* __restrict__ bo,
                       const float* __restrict__ resid,
                       float* __restrict__ y) {
  const int w = threadIdx.x >> 5;
  const int l = threadIdx.x & 31;
  const int hi = l >> 4;
  const int ln = l & 15;
  const int row0 = blockIdx.x * 16;
  const int col0 = blockIdx.y * 512 + w * 64;
  const _Float16* ap = ctx + (size_t)(row0 + ln) * D_MODEL + hi * 8;
  const _Float16* bp0 = WoT + (size_t)(col0 + ln) * D_MODEL + hi * 8;
  v8f acc[4] = {};
  for (int k0 = 0; k0 < D_MODEL; k0 += 32) {
    v16h a = load_frag_f16(ap + k0);
    __builtin_prefetch(ap + k0 + 32, 0, 3);
#pragma unroll
    for (int j = 0; j < 4; ++j) {
      const _Float16* bp = bp0 + (size_t)(j * 16) * D_MODEL + k0;
      v16h b = load_frag_f16(bp);
      __builtin_prefetch(bp + 32, 0, 3);
      acc[j] = wmma16x16x32(a, b, acc[j]);
    }
  }
#pragma unroll
  for (int j = 0; j < 4; ++j) {
    const int col = col0 + j * 16 + ln;
    const float bias = bo[col];
    float* o = y + (size_t)(row0 + hi * 8) * D_MODEL + col;
    const float* rp = resid + (size_t)(row0 + hi * 8) * D_MODEL + col;
#pragma unroll
    for (int r = 0; r < 8; ++r)
      o[(size_t)r * D_MODEL] = acc[j][r] + bias + rp[(size_t)r * D_MODEL];
  }
}

// ---------------------------------------------------------------------------
// Kernel 4: LayerNorm(eps=1e-6) per row of 1024 -> final output.
// ---------------------------------------------------------------------------
__global__ void k_layernorm(const float* __restrict__ y,
                            const float* __restrict__ gamma,
                            const float* __restrict__ beta,
                            float* __restrict__ out) {
  __shared__ float red[256];
  const int t = threadIdx.x;
  const float* p = y + (size_t)blockIdx.x * D_MODEL;
  float vals[4];
  float s = 0.f;
#pragma unroll
  for (int j = 0; j < 4; ++j) {
    vals[j] = p[t + j * 256];
    s += vals[j];
  }
  red[t] = s;
  __syncthreads();
  for (int k = 128; k > 0; k >>= 1) {
    if (t < k) red[t] += red[t + k];
    __syncthreads();
  }
  const float mu = red[0] * (1.0f / D_MODEL);
  __syncthreads();
  s = 0.f;
#pragma unroll
  for (int j = 0; j < 4; ++j) {
    const float d = vals[j] - mu;
    s += d * d;
  }
  red[t] = s;
  __syncthreads();
  for (int k = 128; k > 0; k >>= 1) {
    if (t < k) red[t] += red[t + k];
    __syncthreads();
  }
  const float rstd = rsqrtf(red[0] * (1.0f / D_MODEL) + 1e-6f);
  float* o = out + (size_t)blockIdx.x * D_MODEL;
#pragma unroll
  for (int j = 0; j < 4; ++j) {
    const int ci = t + j * 256;
    o[ci] = (vals[j] - mu) * rstd * gamma[ci] + beta[ci];
  }
}

// ---------------------------------------------------------------------------
extern "C" void kernel_launch(void* const* d_in, const int* in_sizes, int n_in,
                              void* d_out, int out_size, void* d_ws,
                              size_t ws_size, hipStream_t stream) {
  (void)in_sizes; (void)n_in; (void)out_size; (void)ws_size;
  const float* q = (const float*)d_in[0];
  const float* k = (const float*)d_in[1];
  const float* v = (const float*)d_in[2];
  const float* Wq = (const float*)d_in[3];
  const float* bq = (const float*)d_in[4];
  const float* Wk = (const float*)d_in[5];
  const float* bk = (const float*)d_in[6];
  const float* Wv = (const float*)d_in[7];
  const float* bv = (const float*)d_in[8];
  const float* Wo = (const float*)d_in[9];
  const float* bo = (const float*)d_in[10];
  const float* gamma = (const float*)d_in[11];
  const float* beta = (const float*)d_in[12];

  float* out = (float*)d_out;                  // [4096,1024]
  float* attn = out + (size_t)ROWS * D_MODEL;  // [32,2048,2048]

  char* ws = (char*)d_ws;
  const size_t MB = 1024 * 1024;
  _Float16* WqT = (_Float16*)(ws + 0 * MB);   // 2 MB each
  _Float16* WkT = (_Float16*)(ws + 2 * MB);
  _Float16* WvT = (_Float16*)(ws + 4 * MB);
  _Float16* WoT = (_Float16*)(ws + 6 * MB);
  _Float16* qh = (_Float16*)(ws + 8 * MB);    // 8 MB
  _Float16* kh = (_Float16*)(ws + 16 * MB);   // 8 MB
  _Float16* vhT = (_Float16*)(ws + 24 * MB);  // 8 MB
  _Float16* ctx = (_Float16*)(ws + 32 * MB);  // 8 MB
  float* y = (float*)(ws + 40 * MB);          // 16 MB

  const size_t attn_lds = (size_t)16 * LDSW * sizeof(float);  // ~128 KB

  k_wcvt<<<dim3(D_MODEL / 32, D_MODEL / 32, 4), dim3(32, 8), 0, stream>>>(
      Wq, Wk, Wv, Wo, WqT, WkT, WvT, WoT);
  k_qkv<<<dim3(ROWS / 16, D_MODEL / 512, 3), 256, 0, stream>>>(
      q, k, v, WqT, WkT, WvT, bq, bk, bv, qh, kh, vhT);
  k_attn<<<dim3(SEQ / 16, 1, BH), 256, attn_lds, stream>>>(qh, kh, vhT, attn,
                                                           ctx);
  k_proj<<<dim3(ROWS / 16, D_MODEL / 512), 256, 0, stream>>>(ctx, WoT, bo, q, y);
  k_layernorm<<<dim3(ROWS), 256, 0, stream>>>(y, gamma, beta, out);
}
